// Attention_52905407152369
// MI455X (gfx1250) — matrix operand order repaired and checked by reading, not compiled, not run on hardware
//
#include <hip/hip_runtime.h>
#include <hip/hip_bf16.h>

typedef __attribute__((ext_vector_type(16))) __bf16   v16bf;
typedef __attribute__((ext_vector_type(8)))  float    v8f;
typedef __attribute__((ext_vector_type(4)))  unsigned uv4;
typedef int vsi4 __attribute__((vector_size(4 * sizeof(int))));

#define NBATCH 4096
#define CDIM   256
#define TSTEP  32
#define VOC    97
#define VPAD   128

#if __has_builtin(__builtin_amdgcn_global_load_async_to_lds_b128)
#define USE_ASYNC_LDS 1
#endif

// ---------------------------------------------------------------------------
// fp32 -> bf16 weight conversion
// ---------------------------------------------------------------------------
__global__ void cvt_kernel(const float* __restrict__ in, __bf16* __restrict__ out, int n) {
  int i = blockIdx.x * blockDim.x + threadIdx.x;
  if (i < n) out[i] = (__bf16)in[i];
}

__global__ void cvt_pad_kernel(const float* __restrict__ in, __bf16* __restrict__ out,
                               int rows_in, int rows_out, int cols) {
  int i = blockIdx.x * blockDim.x + threadIdx.x;
  if (i < rows_out * cols) {
    int r = i / cols;
    out[i] = (r < rows_in) ? (__bf16)in[i] : (__bf16)0.0f;
  }
}

// ---------------------------------------------------------------------------
// Attention: one block (256 threads, 8 waves) per batch row n.
// Stages enc[t, n, :] (32x256 f32 = 32KB) in LDS — via the CDNA5 async
// global->LDS pipe (ASYNCcnt) when available — then computes additive-
// attention scores, softmax, applied context; writes attn to d_out and the
// bf16 concat([embedded, applied]) A-matrix + bf16 hidden for the GEMMs.
// ---------------------------------------------------------------------------
__global__ void attn_kernel(const int* __restrict__ input, const float* __restrict__ hidden,
                            const float* __restrict__ enc, const float* __restrict__ emb,
                            const float* __restrict__ vat_w, const float* __restrict__ vat_b,
                            float* __restrict__ attn_out, __bf16* __restrict__ xcat,
                            __bf16* __restrict__ hbf) {
  __shared__ float sh_enc[TSTEP * CDIM];
  __shared__ float sh_h[CDIM];
  __shared__ float sh_vw[CDIM];
  __shared__ float sh_sc[TSTEP];
  __shared__ float sh_attn[TSTEP];

  const int n   = blockIdx.x;
  const int tid = threadIdx.x;
  const int lane = tid & 31;
  const int wave = tid >> 5;

#ifdef USE_ASYNC_LDS
  // Async copy: 32 rows x 1KB. Each thread moves 16B per step, 8 steps.
  #pragma unroll
  for (int it = 0; it < 8; ++it) {
    int f = it * 256 + tid;          // 16B chunk index within the tile
    int t = f >> 6;                  // row (64 chunks per 1KB row)
    int c = (f & 63) * 4;            // float column
    const float* g = enc + ((size_t)t * NBATCH + n) * CDIM + c;
    float* l = &sh_enc[t * CDIM + c];
    __builtin_amdgcn_global_load_async_to_lds_b128(
        (__attribute__((address_space(1))) vsi4*)const_cast<float*>(g),
        (__attribute__((address_space(3))) vsi4*)l, 0, 0);
  }
#else
  #pragma unroll 4
  for (int t = 0; t < TSTEP; ++t)
    sh_enc[t * CDIM + tid] = enc[((size_t)t * NBATCH + n) * CDIM + tid];
#endif

  sh_h[tid]  = hidden[(size_t)n * CDIM + tid];
  sh_vw[tid] = vat_w[tid];

#ifdef USE_ASYNC_LDS
#if __has_builtin(__builtin_amdgcn_s_wait_asynccnt)
  __builtin_amdgcn_s_wait_asynccnt(0);
#else
  asm volatile("s_wait_asynccnt 0" ::: "memory");
#endif
#endif
  __syncthreads();

  // scores[t] = sum_c tanh(h[c] + enc[t,c]) * vat_w[c] + vat_b
  #pragma unroll
  for (int q = 0; q < 4; ++q) {
    int t = wave + 8 * q;
    float p = 0.0f;
    #pragma unroll
    for (int j = 0; j < 8; ++j) {
      int c = lane + 32 * j;
      p += tanhf(sh_h[c] + sh_enc[t * CDIM + c]) * sh_vw[c];
    }
    #pragma unroll
    for (int m = 16; m >= 1; m >>= 1) p += __shfl_xor(p, m, 32);
    if (lane == 0) sh_sc[t] = p + vat_b[0];
  }
  __syncthreads();

  // softmax over T=32 in wave 0
  if (tid < 32) {
    float s  = sh_sc[tid];
    float mx = s;
    #pragma unroll
    for (int m = 16; m >= 1; m >>= 1) mx = fmaxf(mx, __shfl_xor(mx, m, 32));
    float e = __expf(s - mx);
    float sum = e;
    #pragma unroll
    for (int m = 16; m >= 1; m >>= 1) sum += __shfl_xor(sum, m, 32);
    float a = e / sum;
    sh_attn[tid] = a;
    attn_out[(size_t)n * TSTEP + tid] = a;
  }
  __syncthreads();

  // applied[c] = sum_t attn[t] * enc[t, c]
  float acc = 0.0f;
  #pragma unroll 4
  for (int t = 0; t < TSTEP; ++t) acc += sh_attn[t] * sh_enc[t * CDIM + tid];

  int vi = input[n];
  xcat[(size_t)n * (2 * CDIM) + tid]        = (__bf16)emb[(size_t)vi * CDIM + tid];
  xcat[(size_t)n * (2 * CDIM) + CDIM + tid] = (__bf16)acc;
  hbf[(size_t)n * CDIM + tid]               = (__bf16)sh_h[tid];
}

// ---------------------------------------------------------------------------
// WMMA bf16 GEMM: D[M x Nout] = A[M x K] * W[Nout x K]^T + bias
// Wave tile: 16 (M) x 32 (N), K-step 32 via v_wmma_f32_16x16x32_bf16.
// Block = 256 threads = 8 waves stacked in M (128 rows/block).
// grid = (Nout/32, M/128). M, Nout, K multiples of 16/32 as required.
// ---------------------------------------------------------------------------
template <bool RELU, bool STORE_BF16>
__global__ void gemm_bf16_kernel(const __bf16* __restrict__ A, const __bf16* __restrict__ W,
                                 const float* __restrict__ bias, int bias_len,
                                 float* __restrict__ Df, __bf16* __restrict__ Db,
                                 int lda, int ldw, int ldd, int K) {
  const int lane = threadIdx.x & 31;
  const int wave = threadIdx.x >> 5;
  const int n0 = blockIdx.x * 32;
  const int m0 = (blockIdx.y * 8 + wave) * 16;

  // A-fragment: lane L holds row m0 + (L&15); K contiguous {akb .. akb+15}
  const int arow = m0 + (lane & 15);
  const int akb  = (lane & 16) ? 16 : 0;
  // B-fragment: lane L holds column n0 + (L&15); K contiguous {bkb .. bkb+15}
  const int bcol = n0 + (lane & 15);
  const int bkb  = (lane & 16) ? 16 : 0;

  const __bf16* ap = A + (size_t)arow * lda + akb;
  const __bf16* bp0 = W + (size_t)bcol * ldw + bkb;
  const __bf16* bp1 = W + (size_t)(bcol + 16) * ldw + bkb;

  v8f acc0 = {};
  v8f acc1 = {};

  for (int k0 = 0; k0 < K; k0 += 32) {
    union { unsigned u[8]; v16bf v; } af;
    const unsigned* a0 = (const unsigned*)(ap + k0);        // K = akb+2j,2j+1
    const unsigned* a1 = (const unsigned*)(ap + k0 + 8);    // K = akb+8+2j
    #pragma unroll
    for (int j = 0; j < 4; ++j) { af.u[j] = a0[j]; af.u[4 + j] = a1[j]; }

    union { uv4 q[2]; v16bf v; } bf0, bf1;
    const uv4* b0 = (const uv4*)(bp0 + k0);
    const uv4* b1 = (const uv4*)(bp1 + k0);
    bf0.q[0] = b0[0]; bf0.q[1] = b0[1];
    bf1.q[0] = b1[0]; bf1.q[1] = b1[1];

    acc0 = __builtin_amdgcn_wmma_f32_16x16x32_bf16(false, af.v, false, bf0.v,
                                                   (short)0, acc0, false, false);
    acc1 = __builtin_amdgcn_wmma_f32_16x16x32_bf16(false, af.v, false, bf1.v,
                                                   (short)0, acc1, false, false);
  }

  // Epilogue. C/D layout: VGPR r -> row m0 + r (+8 for upper lane half); col = lane&15.
  const int mrow = m0 + ((lane & 16) ? 8 : 0);
  #pragma unroll
  for (int half = 0; half < 2; ++half) {
    int col = n0 + (lane & 15) + 16 * half;
    float b = (col < bias_len) ? bias[col] : 0.0f;
    v8f acc = half ? acc1 : acc0;
    #pragma unroll
    for (int r = 0; r < 8; ++r) {
      float v = acc[r] + b;
      if (RELU) v = fmaxf(v, 0.0f);
      size_t off = (size_t)(mrow + r) * ldd + col;
      if (Df) Df[off] = v;
      if (STORE_BF16) Db[off] = (__bf16)v;
    }
  }
}

// ---------------------------------------------------------------------------
// GRU gating (PyTorch order r, z, n): elementwise over N*C
// ---------------------------------------------------------------------------
__global__ void gates_kernel(const float* __restrict__ gi, const float* __restrict__ gh,
                             const float* __restrict__ hidden,
                             float* __restrict__ hnew, __bf16* __restrict__ hnew_bf) {
  int idx = blockIdx.x * blockDim.x + threadIdx.x;
  if (idx >= NBATCH * CDIM) return;
  int n = idx >> 8;
  int c = idx & 255;
  size_t base = (size_t)n * (3 * CDIM);
  float ir = gi[base + c], iz = gi[base + CDIM + c], in_ = gi[base + 2 * CDIM + c];
  float hr = gh[base + c], hz = gh[base + CDIM + c], hn_ = gh[base + 2 * CDIM + c];
  float r = 1.0f / (1.0f + __expf(-(ir + hr)));
  float z = 1.0f / (1.0f + __expf(-(iz + hz)));
  float ng = tanhf(in_ + r * hn_);
  float h = hidden[(size_t)n * CDIM + c];
  float out = (1.0f - z) * ng + z * h;
  hnew[(size_t)n * CDIM + c]    = out;
  hnew_bf[(size_t)n * CDIM + c] = (__bf16)out;
}

// ---------------------------------------------------------------------------
// log-softmax over V=97 (padded 128) logits; one wave per row
// ---------------------------------------------------------------------------
__global__ void lsm_kernel(const float* __restrict__ logits, float* __restrict__ out) {
  int lane = threadIdx.x & 31;
  int wave = threadIdx.x >> 5;
  int n = blockIdx.x * 8 + wave;
  const float* row = logits + (size_t)n * VPAD;
  float s[4];
  float mx = -INFINITY;
  #pragma unroll
  for (int j = 0; j < 4; ++j) {
    int v = lane + 32 * j;
    s[j] = (v < VOC) ? row[v] : -INFINITY;
    mx = fmaxf(mx, s[j]);
  }
  #pragma unroll
  for (int m = 16; m >= 1; m >>= 1) mx = fmaxf(mx, __shfl_xor(mx, m, 32));
  float sum = 0.0f;
  #pragma unroll
  for (int j = 0; j < 4; ++j) {
    int v = lane + 32 * j;
    sum += (v < VOC) ? __expf(s[j] - mx) : 0.0f;
  }
  #pragma unroll
  for (int m = 16; m >= 1; m >>= 1) sum += __shfl_xor(sum, m, 32);
  float lse = mx + __logf(sum);
  float* orow = out + (size_t)n * VOC;
  #pragma unroll
  for (int j = 0; j < 4; ++j) {
    int v = lane + 32 * j;
    if (v < VOC) orow[v] = s[j] - lse;
  }
}

// ---------------------------------------------------------------------------
extern "C" void kernel_launch(void* const* d_in, const int* in_sizes, int n_in,
                              void* d_out, int out_size, void* d_ws, size_t ws_size,
                              hipStream_t stream) {
  const int*   input     = (const int*)d_in[0];
  const float* hidden    = (const float*)d_in[1];
  const float* enc       = (const float*)d_in[2];
  const float* emb       = (const float*)d_in[3];
  const float* combine_w = (const float*)d_in[4];
  const float* combine_b = (const float*)d_in[5];
  const float* vat_w     = (const float*)d_in[6];
  const float* vat_b     = (const float*)d_in[7];
  const float* w_ih      = (const float*)d_in[8];
  const float* w_hh      = (const float*)d_in[9];
  const float* b_ih      = (const float*)d_in[10];
  const float* b_hh      = (const float*)d_in[11];
  const float* out_w     = (const float*)d_in[12];
  const float* out_b     = (const float*)d_in[13];

  float* out_logp = (float*)d_out;                       // [N, V]
  float* out_h    = out_logp + (size_t)NBATCH * VOC;     // [1, N, C]
  float* out_attn = out_h + (size_t)NBATCH * CDIM;       // [N, 1, T]

  // workspace carve-out (256B aligned)
  char* ws = (char*)d_ws;
  size_t off = 0;
  auto carve = [&](size_t bytes) {
    void* p = ws + off;
    off = (off + bytes + 255) & ~(size_t)255;
    return p;
  };
  __bf16* cw_bf    = (__bf16*)carve((size_t)CDIM * 2 * CDIM * 2);        // [256, 512]
  __bf16* wih_bf   = (__bf16*)carve((size_t)3 * CDIM * CDIM * 2);        // [768, 256]
  __bf16* whh_bf   = (__bf16*)carve((size_t)3 * CDIM * CDIM * 2);        // [768, 256]
  __bf16* ow_bf    = (__bf16*)carve((size_t)VPAD * CDIM * 2);            // [128, 256] padded
  __bf16* xcat     = (__bf16*)carve((size_t)NBATCH * 2 * CDIM * 2);      // [N, 512]
  __bf16* x_bf     = (__bf16*)carve((size_t)NBATCH * CDIM * 2);          // [N, 256]
  __bf16* h_bf     = (__bf16*)carve((size_t)NBATCH * CDIM * 2);          // [N, 256]
  __bf16* hnew_bf  = (__bf16*)carve((size_t)NBATCH * CDIM * 2);          // [N, 256]
  float*  gi_f     = (float*)carve((size_t)NBATCH * 3 * CDIM * 4);       // [N, 768]
  float*  gh_f     = (float*)carve((size_t)NBATCH * 3 * CDIM * 4);       // [N, 768]
  float*  logits_f = (float*)carve((size_t)NBATCH * VPAD * 4);           // [N, 128]

  // 1) weight conversion
  cvt_kernel<<<(CDIM * 2 * CDIM + 255) / 256, 256, 0, stream>>>(combine_w, cw_bf, CDIM * 2 * CDIM);
  cvt_kernel<<<(3 * CDIM * CDIM + 255) / 256, 256, 0, stream>>>(w_ih, wih_bf, 3 * CDIM * CDIM);
  cvt_kernel<<<(3 * CDIM * CDIM + 255) / 256, 256, 0, stream>>>(w_hh, whh_bf, 3 * CDIM * CDIM);
  cvt_pad_kernel<<<(VPAD * CDIM + 255) / 256, 256, 0, stream>>>(out_w, ow_bf, VOC, VPAD, CDIM);

  // 2) attention + embedding gather -> xcat bf16, h bf16, attn out
  attn_kernel<<<NBATCH, 256, 0, stream>>>(input, hidden, enc, emb, vat_w, vat_b,
                                          out_attn, xcat, h_bf);

  // 3) combine + relu: x = relu(xcat @ combine_w^T + b)   [N,512]x[256,512]^T
  gemm_bf16_kernel<true, true><<<dim3(CDIM / 32, NBATCH / 128), 256, 0, stream>>>(
      xcat, cw_bf, combine_b, CDIM, nullptr, x_bf, 2 * CDIM, 2 * CDIM, CDIM, 2 * CDIM);

  // 4) gi = x @ W_ih^T + b_ih ; gh = h @ W_hh^T + b_hh    [N,256]x[768,256]^T
  gemm_bf16_kernel<false, false><<<dim3(3 * CDIM / 32, NBATCH / 128), 256, 0, stream>>>(
      x_bf, wih_bf, b_ih, 3 * CDIM, gi_f, nullptr, CDIM, CDIM, 3 * CDIM, CDIM);
  gemm_bf16_kernel<false, false><<<dim3(3 * CDIM / 32, NBATCH / 128), 256, 0, stream>>>(
      h_bf, whh_bf, b_hh, 3 * CDIM, gh_f, nullptr, CDIM, CDIM, 3 * CDIM, CDIM);

  // 5) GRU gating -> h_new (f32 output + bf16 for final GEMM)
  gates_kernel<<<(NBATCH * CDIM + 255) / 256, 256, 0, stream>>>(gi_f, gh_f, hidden,
                                                                out_h, hnew_bf);

  // 6) logits = h_new @ out_w^T + out_b  (padded to 128 cols)
  gemm_bf16_kernel<false, false><<<dim3(VPAD / 32, NBATCH / 128), 256, 0, stream>>>(
      hnew_bf, ow_bf, out_b, VOC, logits_f, nullptr, CDIM, CDIM, VPAD, CDIM);

  // 7) log-softmax over the 97 valid columns
  lsm_kernel<<<NBATCH / 8, 256, 0, stream>>>(logits_f, out_logp);
}
